// Shift_tcn_layer_69973607186685
// MI455X (gfx1250) — matrix operand order repaired and checked by reading, not compiled, not run on hardware
//
#include <hip/hip_runtime.h>

typedef __attribute__((ext_vector_type(16))) _Float16 v16h;
typedef __attribute__((ext_vector_type(8)))  _Float16 v8h;
typedef __attribute__((ext_vector_type(4)))  _Float16 v4h;
typedef __attribute__((ext_vector_type(8)))  float    v8f;
typedef __attribute__((ext_vector_type(4)))  int      v4i;

#define NROWS 131072   // N*T = 128*1024
#define TDIM  1024
#define CIN   256
#define COUT  256
#define NT_INV (1.0f / (128.0f * 1024.0f))
#define EPS 1e-5f

// ---- CDNA5 async global->LDS path (ASYNCcnt), guarded so compile can't break
#define GAS __attribute__((address_space(1)))
#define LAS __attribute__((address_space(3)))

#if defined(__gfx1250__) && __has_builtin(__builtin_amdgcn_global_load_async_to_lds_b128)
#define USE_ASYNC_LDS 1
#else
#define USE_ASYNC_LDS 0
#endif

__device__ __forceinline__ void async_cp16(_Float16* lds, const _Float16* glb) {
#if USE_ASYNC_LDS
  __builtin_amdgcn_global_load_async_to_lds_b128(
      (GAS v4i*)const_cast<_Float16*>(glb), (LAS v4i*)lds, 0, 0);
#else
  *(v8h*)lds = *(const v8h*)glb;
#endif
}

#if USE_ASYNC_LDS
#if __has_builtin(__builtin_amdgcn_s_wait_asynccnt)
#define WAIT_ASYNC(n) __builtin_amdgcn_s_wait_asynccnt(n)
#else
#define WAIT_ASYNC(n) asm volatile("s_wait_asynccnt %0" ::"i"(n) : "memory")
#endif
#else
#define WAIT_ASYNC(n) ((void)0)
#endif

// ---------------- K1: BN1 stats (per input channel partial sums) -------------
__global__ __launch_bounds__(256) void bn1_stats(const float* __restrict__ x,
                                                 float* __restrict__ psum,
                                                 float* __restrict__ psq) {
  int c = threadIdx.x;
  float s = 0.f, q = 0.f;
  for (long r = blockIdx.x; r < NROWS; r += gridDim.x) {
    float v = x[r * CIN + c];
    s += v; q += v * v;
  }
  psum[(long)blockIdx.x * CIN + c] = s;
  psq [(long)blockIdx.x * CIN + c] = q;
}

// ---------------- K2: finalize BN1 affine + both shift parameter sets --------
__global__ __launch_bounds__(256) void finalize1(
    const float* __restrict__ psum, const float* __restrict__ psq,
    const float* __restrict__ gamma, const float* __restrict__ beta,
    const float* __restrict__ s_in, const float* __restrict__ s_out,
    float* a1, float* b1,
    int* dI, float* wi0, float* wi1,
    int* dO, float* wo0, float* wo1) {
  int c = threadIdx.x;
  float s = 0.f, q = 0.f;
  for (int b = 0; b < 512; ++b) { s += psum[b * CIN + c]; q += psq[b * CIN + c]; }
  float m = s * NT_INV;
  float var = q * NT_INV - m * m;
  float rstd = rsqrtf(var + EPS);
  float A = gamma[c] * rstd;
  a1[c] = A; b1[c] = beta[c] - m * A;
  float si = s_in[c], di = floorf(si);
  dI[c] = (int)di; wi1[c] = si - di; wi0[c] = 1.f - (si - di);
  float so = s_out[c], dos = floorf(so);
  dO[c] = (int)dos; wo1[c] = so - dos; wo0[c] = 1.f - (so - dos);
}

// ---------------- K2b: convert conv weights to f16 ---------------------------
__global__ __launch_bounds__(256) void convw(const float* __restrict__ w,
                                             _Float16* __restrict__ wh) {
  long i = (long)blockIdx.x * 256 + threadIdx.x;
  wh[i] = (_Float16)w[i];
}

// ---------------- K3: BN1 apply + temporal shift (in) -> h f16 ---------------
__global__ __launch_bounds__(256) void shift_in_apply(
    const float* __restrict__ x,
    const float* __restrict__ a1, const float* __restrict__ b1,
    const int* __restrict__ dsh, const float* __restrict__ w0a,
    const float* __restrict__ w1a, _Float16* __restrict__ h) {
  int tid = threadIdx.x;
  long row = (long)blockIdx.x * 4 + (tid >> 6);
  int c0 = (tid & 63) * 4;
  int t = (int)(row & (TDIM - 1));
  long rm1 = (t > 0) ? row - 1 : row;
  long rp1 = (t < TDIM - 1) ? row + 1 : row;
  float4 xm = *(const float4*)(x + rm1 * CIN + c0);
  float4 x0 = *(const float4*)(x + row * CIN + c0);
  float4 xp = *(const float4*)(x + rp1 * CIN + c0);
  const float* xmv = (const float*)&xm;
  const float* x0v = (const float*)&x0;
  const float* xpv = (const float*)&xp;
  v4h out;
#pragma unroll
  for (int j = 0; j < 4; ++j) {
    int c = c0 + j;
    int d = dsh[c];
    float A = a1[c], B = b1[c], w0 = w0a[c], w1 = w1a[c];
    float tap0 = (d < 0) ? xmv[j] : x0v[j];
    float tap1 = (d < 0) ? x0v[j] : xpv[j];
    float v0 = (t + d >= 0) ? 1.f : 0.f;
    float v1 = (t + d + 1 <= TDIM - 1) ? 1.f : 0.f;
    float val = w0 * v0 * (A * tap0 + B) + w1 * v1 * (A * tap1 + B);
    out[j] = (_Float16)val;
  }
  *(v4h*)(h + row * CIN + c0) = out;
}

// ---------------- K4: WMMA GEMM (h * w^T) + bias + ReLU -> z f16 -------------
// Block tile: M=256, N=64, K-steps of 32. 256 threads = 8 waves.
// A tiles double-buffered in LDS, filled by async global->LDS (ASYNCcnt),
// prefetching K-step k+1 while WMMAs consume step k.
__global__ __launch_bounds__(256) void gemm_bias_relu(
    const _Float16* __restrict__ h, const _Float16* __restrict__ wh,
    const float* __restrict__ bias, _Float16* __restrict__ z) {
  __shared__ _Float16 As[2][256 * 32];  // 2 x 16 KB double buffer
  __shared__ _Float16 Bs[64 * 256];     // 32 KB: full-K weight panel (64 out ch)
  int tid = threadIdx.x;
  long row0 = (long)blockIdx.x * 256;
  int n0 = blockIdx.y * 64;

  // Stage B panel: 64 rows x 256 cols f16 (row = output channel, col = k)
#pragma unroll
  for (int i = 0; i < 8; ++i) {
    int chunk = tid + i * 256;            // 2048 chunks of 8 f16
    int r = chunk >> 5, cc = chunk & 31;
    async_cp16(Bs + r * 256 + cc * 8, wh + (long)(n0 + r) * CIN + cc * 8);
  }
  // Prologue: A tile for K-step 0 into buffer 0
#pragma unroll
  for (int i = 0; i < 4; ++i) {
    int chunk = tid + i * 256;            // 1024 chunks of 8 f16
    int r = chunk >> 2, cc = chunk & 3;
    async_cp16(As[0] + r * 32 + cc * 8, h + (row0 + r) * CIN + cc * 8);
  }

  int wv = tid >> 5, lane = tid & 31;
  int halfsel = lane >> 4, l16 = lane & 15;
  v8f acc[2][4] = {};

#pragma unroll
  for (int ks = 0; ks < 8; ++ks) {
    int cur = ks & 1;
    if (ks < 7) {
      // Prefetch next A tile into the other buffer while this one computes.
#pragma unroll
      for (int i = 0; i < 4; ++i) {
        int chunk = tid + i * 256;
        int r = chunk >> 2, cc = chunk & 3;
        async_cp16(As[cur ^ 1] + r * 32 + cc * 8,
                   h + (row0 + r) * CIN + (ks + 1) * 32 + cc * 8);
      }
      WAIT_ASYNC(4);   // oldest 4 (current tile, + B panel on ks==0) complete
    } else {
      WAIT_ASYNC(0);
    }
    __syncthreads();

    const _Float16* Ab = As[cur];
    // B fragments: lane holds column n=l16; K chunk halfsel*16 .. +16 (contig)
    v16h bf[4];
#pragma unroll
    for (int nt = 0; nt < 4; ++nt) {
      const _Float16* bp = Bs + (nt * 16 + l16) * 256 + ks * 32 + halfsel * 16;
      v8h blo = *(const v8h*)bp;
      v8h bhi = *(const v8h*)(bp + 8);
      bf[nt] = __builtin_shufflevector(blo, bhi, 0, 1, 2, 3, 4, 5, 6, 7,
                                       8, 9, 10, 11, 12, 13, 14, 15);
    }
    // A fragments: lane holds row m=l16; K chunk halfsel*16 .. +16 (contig)
#pragma unroll
    for (int mt = 0; mt < 2; ++mt) {
      const _Float16* ap = Ab + (wv * 32 + mt * 16 + l16) * 32 + halfsel * 16;
      v8h alo = *(const v8h*)ap;
      v8h ahi = *(const v8h*)(ap + 8);
      v16h af = __builtin_shufflevector(alo, ahi, 0, 1, 2, 3, 4, 5, 6, 7,
                                        8, 9, 10, 11, 12, 13, 14, 15);
#pragma unroll
      for (int nt = 0; nt < 4; ++nt) {
        acc[mt][nt] = __builtin_amdgcn_wmma_f32_16x16x32_f16(
            false, af, false, bf[nt], (short)0, acc[mt][nt], false, false);
      }
    }
    __syncthreads();  // protect buffer being prefetched next iteration
  }

  // Epilogue: C layout — lanes 0-15: N=lane, M=v; lanes 16-31: N=lane-16, M=v+8
#pragma unroll
  for (int mt = 0; mt < 2; ++mt) {
    long gm0 = row0 + wv * 32 + mt * 16 + halfsel * 8;
#pragma unroll
    for (int nt = 0; nt < 4; ++nt) {
      int go = n0 + nt * 16 + l16;
      float bv = bias[go];
      v8f a = acc[mt][nt];
#pragma unroll
      for (int v = 0; v < 8; ++v) {
        float val = a[v] + bv;
        val = val > 0.f ? val : 0.f;
        z[(gm0 + v) * COUT + go] = (_Float16)val;
      }
    }
  }
}

// ---------------- K5: shift_out on the fly + BN2 partial stats ---------------
__global__ __launch_bounds__(256) void bn2_stats(
    const _Float16* __restrict__ z, const int* __restrict__ dsh,
    const float* __restrict__ w0a, const float* __restrict__ w1a,
    float* __restrict__ psum, float* __restrict__ psq) {
  int c = threadIdx.x;
  int d = dsh[c];
  float w0 = w0a[c], w1 = w1a[c];
  float s = 0.f, q = 0.f;
  for (long row = blockIdx.x; row < NROWS; row += gridDim.x) {
    int t = (int)(row & (TDIM - 1));
    long rm1 = (t > 0) ? row - 1 : row;
    long rp1 = (t < TDIM - 1) ? row + 1 : row;
    float zm = (float)z[rm1 * COUT + c];
    float z0 = (float)z[row * COUT + c];
    float zp = (float)z[rp1 * COUT + c];
    float tap0 = (d < 0) ? zm : z0;
    float tap1 = (d < 0) ? z0 : zp;
    float v0 = (t + d >= 0) ? 1.f : 0.f;
    float v1 = (t + d + 1 <= TDIM - 1) ? 1.f : 0.f;
    float y = w0 * v0 * tap0 + w1 * v1 * tap1;
    s += y; q += y * y;
  }
  psum[(long)blockIdx.x * COUT + c] = s;
  psq [(long)blockIdx.x * COUT + c] = q;
}

// ---------------- K6: finalize BN2 affine ------------------------------------
__global__ __launch_bounds__(256) void finalize2(
    const float* __restrict__ psum, const float* __restrict__ psq,
    const float* __restrict__ gamma, const float* __restrict__ beta,
    float* a2, float* b2) {
  int c = threadIdx.x;
  float s = 0.f, q = 0.f;
  for (int b = 0; b < 512; ++b) { s += psum[b * COUT + c]; q += psq[b * COUT + c]; }
  float m = s * NT_INV;
  float var = q * NT_INV - m * m;
  float rstd = rsqrtf(var + EPS);
  float A = gamma[c] * rstd;
  a2[c] = A; b2[c] = beta[c] - m * A;
}

// ---------------- K7: shift_out + BN2 apply -> f32 output --------------------
__global__ __launch_bounds__(256) void shift_out_apply(
    const _Float16* __restrict__ z, const int* __restrict__ dsh,
    const float* __restrict__ w0a, const float* __restrict__ w1a,
    const float* __restrict__ a2, const float* __restrict__ b2,
    float* __restrict__ out) {
  int tid = threadIdx.x;
  long row = (long)blockIdx.x * 4 + (tid >> 6);
  int c0 = (tid & 63) * 4;
  int t = (int)(row & (TDIM - 1));
  long rm1 = (t > 0) ? row - 1 : row;
  long rp1 = (t < TDIM - 1) ? row + 1 : row;
  v4h zm4 = *(const v4h*)(z + rm1 * COUT + c0);
  v4h z04 = *(const v4h*)(z + row * COUT + c0);
  v4h zp4 = *(const v4h*)(z + rp1 * COUT + c0);
  float4 o;
  float* ov = (float*)&o;
#pragma unroll
  for (int j = 0; j < 4; ++j) {
    int c = c0 + j;
    int d = dsh[c];
    float w0 = w0a[c], w1 = w1a[c];
    float zm = (float)zm4[j], z0 = (float)z04[j], zp = (float)zp4[j];
    float tap0 = (d < 0) ? zm : z0;
    float tap1 = (d < 0) ? z0 : zp;
    float v0 = (t + d >= 0) ? 1.f : 0.f;
    float v1 = (t + d + 1 <= TDIM - 1) ? 1.f : 0.f;
    float y = w0 * v0 * tap0 + w1 * v1 * tap1;
    ov[j] = a2[c] * y + b2[c];
  }
  *(float4*)(out + row * COUT + c0) = o;
}

// ---------------- launch -----------------------------------------------------
extern "C" void kernel_launch(void* const* d_in, const int* in_sizes, int n_in,
                              void* d_out, int out_size, void* d_ws, size_t ws_size,
                              hipStream_t stream) {
  const float* x     = (const float*)d_in[0];
  const float* g1    = (const float*)d_in[1];
  const float* be1   = (const float*)d_in[2];
  const float* s_in  = (const float*)d_in[3];
  const float* w     = (const float*)d_in[4];
  const float* bias  = (const float*)d_in[5];
  const float* s_out = (const float*)d_in[6];
  const float* g2    = (const float*)d_in[7];
  const float* be2   = (const float*)d_in[8];
  float* out = (float*)d_out;

  char* ws = (char*)d_ws;
  const size_t HBYTES = (size_t)NROWS * CIN * sizeof(_Float16);   // 67 MB
  const size_t ZBYTES = (size_t)NROWS * COUT * sizeof(_Float16);  // 67 MB
  const size_t PBYTES = (size_t)512 * 256 * sizeof(float);        // 512 KB each
  size_t off = 0;
  _Float16* h   = (_Float16*)(ws + off); off += HBYTES;
  _Float16* z   = (_Float16*)(ws + off); off += ZBYTES;
  _Float16* wh  = (_Float16*)(ws + off); off += (size_t)COUT * CIN * sizeof(_Float16);
  float* ps1 = (float*)(ws + off); off += PBYTES;
  float* pq1 = (float*)(ws + off); off += PBYTES;
  float* ps2 = (float*)(ws + off); off += PBYTES;
  float* pq2 = (float*)(ws + off); off += PBYTES;
  float* a1  = (float*)(ws + off); off += 1024;
  float* b1  = (float*)(ws + off); off += 1024;
  float* a2  = (float*)(ws + off); off += 1024;
  float* b2  = (float*)(ws + off); off += 1024;
  int*   dI  = (int*)  (ws + off); off += 1024;
  float* wi0 = (float*)(ws + off); off += 1024;
  float* wi1 = (float*)(ws + off); off += 1024;
  int*   dO  = (int*)  (ws + off); off += 1024;
  float* wo0 = (float*)(ws + off); off += 1024;
  float* wo1 = (float*)(ws + off); off += 1024;

  bn1_stats<<<512, 256, 0, stream>>>(x, ps1, pq1);
  finalize1<<<1, 256, 0, stream>>>(ps1, pq1, g1, be1, s_in, s_out,
                                   a1, b1, dI, wi0, wi1, dO, wo0, wo1);
  convw<<<(COUT * CIN) / 256, 256, 0, stream>>>(w, wh);
  shift_in_apply<<<NROWS / 4, 256, 0, stream>>>(x, a1, b1, dI, wi0, wi1, h);
  dim3 gg(NROWS / 256, COUT / 64);
  gemm_bias_relu<<<gg, 256, 0, stream>>>(h, wh, bias, z);
  bn2_stats<<<512, 256, 0, stream>>>(z, dO, wo0, wo1, ps2, pq2);
  finalize2<<<1, 256, 0, stream>>>(ps2, pq2, g2, be2, a2, b2);
  shift_out_apply<<<NROWS / 4, 256, 0, stream>>>(z, dO, wo0, wo1, a2, b2, out);
}
